// NeRF_28656021799302
// MI455X (gfx1250) — matrix operand order repaired and checked by reading, not compiled, not run on hardware
//
#include <hip/hip_runtime.h>
#include <hip/hip_bf16.h>
#include <math.h>

// ---------------- constants from the reference ----------------
#define N_RAYS_C   1024
#define N_COARSE_C 64
#define N_ALL_C    192      // 64 coarse + 128 fine
#define NEAR_T     2.0f
#define FAR_T      6.0f

typedef __attribute__((ext_vector_type(16))) _Float16 v16h;
typedef __attribute__((ext_vector_type(8)))  _Float16 v8h;
typedef __attribute__((ext_vector_type(8)))  float    v8f;

// ---------------- WMMA fragment loaders (16x16x32 f16; contiguous k order, same for A and B) -------
// A-matrix 16x32 (MxK), 16-bit: lane<16 holds row M=lane, halves 0..15 = K
// [kb+hi*16 .. +16)  (hi = lane>>4), matching the B fragment.
__device__ __forceinline__ v16h load_a_frag(const _Float16* src, int ld,
                                            int rowBase, int kBase, int lane) {
  const int r  = rowBase + (lane & 15);
  const int hi = lane >> 4;
  const _Float16* p = src + r * ld + kBase + hi * 16;
  v8h lo  = *(const v8h*)(p);
  v8h hiv = *(const v8h*)(p + 8);
  v16h out;
#pragma unroll
  for (int i = 0; i < 8; ++i) { out[i] = lo[i]; out[i + 8] = hiv[i]; }
  return out;
}

// B-matrix 32x16 (KxN), this code's k order (same as A): lane<16 -> N=lane, K=h; lane>=16 -> K=16+h.
// Weights stored TRANSPOSED as Wt[dout_pad][K_pad] so this is one contiguous
// 32-byte load per lane.
__device__ __forceinline__ v16h load_b_frag(const _Float16* Wt, int Kp,
                                            int nBase, int kBase, int lane) {
  const int n  = nBase + (lane & 15);
  const int hi = lane >> 4;
  return *(const v16h*)(Wt + (size_t)n * Kp + kBase + hi * 16);
}

__device__ __forceinline__ v8f wmma_f16(v16h a, v16h b, v8f c) {
  return __builtin_amdgcn_wmma_f32_16x16x32_f16(false, a, false, b, (short)0, c,
                                                false, false);
}

__device__ __forceinline__ float apply_act(float x, int ACT) {
  if (ACT == 1) return fmaxf(x, 0.f);
  if (ACT == 2) return 1.f / (1.f + expf(-x));
  return x;
}

// ---------------- generic fused MLP layer (tile GEMM + bias + act) ---------
// Block = 256 threads = 8 waves, 64 rows per block.
// K-step outer loop: A fragments (4 row tiles) and B fragments (CTW col tiles
// per wave) are each loaded ONCE per K-step and reused across all 4*CTW WMMAs.
// ACT: 0=none, 1=relu, 2=sigmoid.
template<int KP, int KSPLIT, int DOUTPAD, int ACT>
__device__ __forceinline__ void mlp_layer(
    const _Float16* __restrict__ Wt, const float* __restrict__ bias, int doutReal,
    const _Float16* srcA, int ldA,          // K in [0, KSPLIT)
    const _Float16* srcB, int ldB,          // K in [KSPLIT, KP)   (concat region)
    _Float16* dstLds, int ldDst,            // LDS f16 destination (or nullptr)
    float* gOut, int gLd, int gRow0)        // global f32 destination (or nullptr)
{
  constexpr int NT = DOUTPAD / 16;
  const int lane = threadIdx.x & 31;
  const int wave = threadIdx.x >> 5;

  if constexpr (NT >= 8) {
    constexpr int CTW = NT / 8;             // col tiles per wave (1 or 2)
    v8f accs[CTW][4];
#pragma unroll
    for (int c2 = 0; c2 < CTW; ++c2)
#pragma unroll
      for (int rt = 0; rt < 4; ++rt)
        accs[c2][rt] = v8f{0.f, 0.f, 0.f, 0.f, 0.f, 0.f, 0.f, 0.f};
#pragma unroll
    for (int ks = 0; ks < KP / 32; ++ks) {
      const int kb = ks * 32;
      const _Float16* s = (kb < KSPLIT) ? srcA : srcB;
      const int ld      = (kb < KSPLIT) ? ldA  : ldB;
      const int kk      = (kb < KSPLIT) ? kb   : (kb - KSPLIT);
      v16h a[4];
#pragma unroll
      for (int rt = 0; rt < 4; ++rt) a[rt] = load_a_frag(s, ld, rt * 16, kk, lane);
#pragma unroll
      for (int c2 = 0; c2 < CTW; ++c2) {
        const int ct = wave + c2 * 8;
        v16h b = load_b_frag(Wt, KP, ct * 16, kb, lane);
#pragma unroll
        for (int rt = 0; rt < 4; ++rt)
          accs[c2][rt] = wmma_f16(a[rt], b, accs[c2][rt]);
      }
    }
    __syncthreads();                        // all reads done before in-place write
#pragma unroll
    for (int c2 = 0; c2 < CTW; ++c2) {
      const int ct = wave + c2 * 8;
#pragma unroll
      for (int rt = 0; rt < 4; ++rt) {
#pragma unroll
        for (int v = 0; v < 8; ++v) {
          // C/D layout: VGPR v -> M = v (lanes 0-15) / 8+v (lanes 16-31), N = lane&15
          const int r = rt * 16 + ((lane < 16) ? v : v + 8);
          const int c = ct * 16 + (lane & 15);
          float x = apply_act(accs[c2][rt][v] + ((c < doutReal) ? bias[c] : 0.f), ACT);
          if (dstLds) dstLds[r * ldDst + c] = (_Float16)x;
          if (gOut && c < doutReal) gOut[(size_t)(gRow0 + r) * gLd + c] = x;
        }
      }
    }
    __syncthreads();
  } else {
    // Narrow head (DOUTPAD == 16): waves 0..3 take row tile rt = wave, ct = 0.
    v8f acc = v8f{0.f, 0.f, 0.f, 0.f, 0.f, 0.f, 0.f, 0.f};
    const bool active = (wave < 4);         // wave-uniform -> EXEC stays full
    if (active) {
#pragma unroll
      for (int ks = 0; ks < KP / 32; ++ks) {
        const int kb = ks * 32;
        const _Float16* s = (kb < KSPLIT) ? srcA : srcB;
        const int ld      = (kb < KSPLIT) ? ldA  : ldB;
        const int kk      = (kb < KSPLIT) ? kb   : (kb - KSPLIT);
        v16h a = load_a_frag(s, ld, wave * 16, kk, lane);
        v16h b = load_b_frag(Wt, KP, 0, kb, lane);
        acc = wmma_f16(a, b, acc);
      }
    }
    __syncthreads();
    if (active) {
#pragma unroll
      for (int v = 0; v < 8; ++v) {
        const int r = wave * 16 + ((lane < 16) ? v : v + 8);
        const int c = lane & 15;
        float x = apply_act(acc[v] + ((c < doutReal) ? bias[c] : 0.f), ACT);
        if (dstLds) dstLds[r * ldDst + c] = (_Float16)x;
        if (gOut && c < doutReal) gOut[(size_t)(gRow0 + r) * gLd + c] = x;
      }
    }
    __syncthreads();
  }
}

// ---------------- positional encoding helper ------------------------------
// layout: [x(3), sin(2^i * x_d) for i<deg,d<3, cos(...)] then zero-pad.
__device__ __forceinline__ float enc_val(const float* p, int c, int deg) {
  const int nf = 3 * deg;
  if (c < 3) return p[c];
  if (c < 3 + nf)     { int i = c - 3;      float sc = (float)(1 << (i / 3)); return sinf(sc * p[i % 3]); }
  if (c < 3 + 2 * nf) { int i = c - 3 - nf; float sc = (float)(1 << (i / 3)); return cosf(sc * p[i % 3]); }
  return 0.f;
}

// ---------------- parameter bundle ----------------------------------------
struct MlpParams {
  const _Float16* wt[8];  const float* bt[8];
  const _Float16* wsig;   const float* bsig;
  const _Float16* wbot;   const float* bbot;
  const _Float16* wview;  const float* bview;
  const _Float16* wrgb;   const float* brgb;
};

// ---------------- fully fused NeRF MLP ------------------------------------
__global__ __launch_bounds__(256) void fused_nerf_mlp(
    MlpParams P,
    const float* __restrict__ origins, const float* __restrict__ dirs,
    const _Float16* __restrict__ vdenc, const float* __restrict__ t_all, int S,
    float* __restrict__ sigmaOut, float* __restrict__ rgbOut, int needRgb)
{
  __shared__ __align__(128) _Float16 sAct[64 * 256];  // activations (in-place)
  __shared__ __align__(128) _Float16 sEnc[64 * 64];   // posenc (63 + 1 pad)
  __shared__ __align__(128) _Float16 sVd [64 * 32];   // viewenc (27 + 5 pad)

  const int row0 = blockIdx.x * 64;
  const int tid  = threadIdx.x;
  {
    const int r = tid & 63, q = tid >> 6;            // 4 threads per row
    const int grow = row0 + r;
    const int ray = grow / S, s = grow % S;
    const float t = t_all ? t_all[grow]
                          : (NEAR_T + (FAR_T - NEAR_T) * (float)s / (float)(S - 1));
    float p[3];
#pragma unroll
    for (int d = 0; d < 3; ++d) p[d] = origins[ray * 3 + d] + t * dirs[ray * 3 + d];
#pragma unroll
    for (int jj = 0; jj < 16; ++jj) {
      const int c = q * 16 + jj;
      sEnc[r * 64 + c] = (_Float16)enc_val(p, c, 10);
    }
#pragma unroll
    for (int jj = 0; jj < 8; ++jj)
      sVd[r * 32 + q * 8 + jj] = vdenc[ray * 32 + q * 8 + jj];
  }
  __syncthreads();

  // trunk: 63->256, 4x 256->256, skip-concat 319->256, 2x 256->256 (relu)
  mlp_layer< 64, 64,256,1>(P.wt[0], P.bt[0], 256, sEnc,  64, sEnc,  64, sAct, 256, nullptr, 0, row0);
  mlp_layer<256,256,256,1>(P.wt[1], P.bt[1], 256, sAct, 256, sAct, 256, sAct, 256, nullptr, 0, row0);
  mlp_layer<256,256,256,1>(P.wt[2], P.bt[2], 256, sAct, 256, sAct, 256, sAct, 256, nullptr, 0, row0);
  mlp_layer<256,256,256,1>(P.wt[3], P.bt[3], 256, sAct, 256, sAct, 256, sAct, 256, nullptr, 0, row0);
  mlp_layer<256,256,256,1>(P.wt[4], P.bt[4], 256, sAct, 256, sAct, 256, sAct, 256, nullptr, 0, row0);
  mlp_layer<320,256,256,1>(P.wt[5], P.bt[5], 256, sAct, 256, sEnc,  64, sAct, 256, nullptr, 0, row0); // concat enc
  mlp_layer<256,256,256,1>(P.wt[6], P.bt[6], 256, sAct, 256, sAct, 256, sAct, 256, nullptr, 0, row0);
  mlp_layer<256,256,256,1>(P.wt[7], P.bt[7], 256, sAct, 256, sAct, 256, sAct, 256, nullptr, 0, row0);

  // sigma head: 256 -> 1 (relu), before bottleneck overwrites sAct
  mlp_layer<256,256, 16,1>(P.wsig, P.bsig, 1, sAct, 256, sAct, 256, nullptr, 0, sigmaOut, 1, row0);

  if (needRgb) {
    // bottleneck 256->256 (linear, in place)
    mlp_layer<256,256,256,0>(P.wbot,  P.bbot,  256, sAct, 256, sAct, 256, sAct, 256, nullptr, 0, row0);
    // view: concat(bottleneck, vdenc) 283->128 (relu), writes cols 0..127
    mlp_layer<288,256,128,1>(P.wview, P.bview, 128, sAct, 256, sVd,   32, sAct, 256, nullptr, 0, row0);
    // rgb: 128 -> 3 (sigmoid) to global
    mlp_layer<128,128, 16,2>(P.wrgb,  P.brgb,    3, sAct, 256, sAct, 256, nullptr, 0, rgbOut, 3, row0);
  }
}

// ---------------- weight convert: f32 [din,dout] -> f16 T [doutPad,Kp] -----
__global__ void convert_weight(const float* __restrict__ W, _Float16* __restrict__ Wt,
                               int din, int dout, int Kp, int doutPad) {
  const int idx = blockIdx.x * blockDim.x + threadIdx.x;
  const int total = Kp * doutPad;
  if (idx >= total) return;
  const int n = idx / Kp, k = idx % Kp;
  const float v = (k < din && n < dout) ? W[(size_t)k * dout + n] : 0.f;
  Wt[(size_t)n * Kp + k] = (_Float16)v;
}

// ---------------- viewdir encoding (per ray, deg 4, padded to 32) ----------
__global__ void vdenc_kernel(const float* __restrict__ viewdirs, _Float16* __restrict__ out) {
  const int ray = blockIdx.x * blockDim.x + threadIdx.x;
  if (ray >= N_RAYS_C) return;
  float v[3] = { viewdirs[ray * 3], viewdirs[ray * 3 + 1], viewdirs[ray * 3 + 2] };
  for (int c = 0; c < 32; ++c) out[ray * 32 + c] = (_Float16)enc_val(v, c, 4);
}

// ---------------- coarse volumetric weights --------------------------------
__global__ void render_coarse_weights(const float* __restrict__ sigma,
                                      const float* __restrict__ viewdirs,
                                      float* __restrict__ wts) {
  const int ray = blockIdx.x * blockDim.x + threadIdx.x;
  if (ray >= N_RAYS_C) return;
  const float dx = viewdirs[ray * 3], dy = viewdirs[ray * 3 + 1], dz = viewdirs[ray * 3 + 2];
  const float dn = sqrtf(dx * dx + dy * dy + dz * dz);
  const float dt = (FAR_T - NEAR_T) / 63.f;
  float trans = 1.f;
  for (int i = 0; i < 64; ++i) {
    const float dist  = ((i < 63) ? dt : 1e10f) * dn;
    const float alpha = 1.f - expf(-sigma[ray * 64 + i] * dist);
    wts[ray * 64 + i] = alpha * trans;
    trans *= (1.f - alpha + 1e-10f);
  }
}

// ---------------- inverse-CDF fine sampling + sorted merge -----------------
__global__ void pdf_merge_kernel(const float* __restrict__ wts, float* __restrict__ tAll) {
  const int ray = blockIdx.x * blockDim.x + threadIdx.x;
  if (ray >= N_RAYS_C) return;
  const float dt = (FAR_T - NEAR_T) / 63.f;
  float w[62];
  float wsum = 0.f;
  for (int j = 0; j < 62; ++j) { w[j] = wts[ray * 64 + 1 + j]; wsum += w[j]; }
  const float pad = fmaxf(0.f, 1e-5f - wsum);
  for (int j = 0; j < 62; ++j) w[j] += pad / 62.f;
  wsum += pad;
  float cdf[63];
  cdf[0] = 0.f;
  float acc = 0.f;
  for (int j = 0; j < 61; ++j) { acc += w[j] / wsum; cdf[j + 1] = fminf(1.f, acc); }
  cdf[62] = 1.f;
  float ts[128];
  const float du = (1.f - 1.1920929e-7f) / 127.f;
  for (int k = 0; k < 128; ++k) {
    const float u = k * du;
    int lo = 0, hi = 63;                                // searchsorted(side='right')
    while (lo < hi) { int mid = (lo + hi) >> 1; if (cdf[mid] <= u) lo = mid + 1; else hi = mid; }
    int idx = lo - 1;
    int i0 = idx < 0 ? 0 : (idx > 62 ? 62 : idx);
    int i1 = (idx + 1) > 62 ? 62 : (idx + 1);
    const float c0 = cdf[i0], c1 = cdf[i1];
    const float b0 = NEAR_T + dt * (i0 + 0.5f);
    const float b1 = NEAR_T + dt * (i1 + 0.5f);
    float tt = (u - c0) / (c1 - c0);
    if (!(tt == tt)) tt = 0.f;
    tt = fminf(fmaxf(tt, 0.f), 1.f);
    ts[k] = b0 + tt * (b1 - b0);
  }
  // merge sorted linspace(64) with sorted fine samples(128) == jnp.sort(concat)
  int ia = 0, ib = 0;
  for (int m = 0; m < N_ALL_C; ++m) {
    const float va = (ia < 64)  ? (NEAR_T + dt * ia) : 3.4e38f;
    const float vb = (ib < 128) ? ts[ib]             : 3.4e38f;
    if (va <= vb) { tAll[ray * N_ALL_C + m] = va; ++ia; }
    else          { tAll[ray * N_ALL_C + m] = vb; ++ib; }
  }
}

// ---------------- fine compositing -> d_out --------------------------------
__global__ void render_fine_kernel(const float* __restrict__ sigma,
                                   const float* __restrict__ rgb,
                                   const float* __restrict__ tAll,
                                   const float* __restrict__ viewdirs,
                                   float* __restrict__ out) {
  const int ray = blockIdx.x * blockDim.x + threadIdx.x;
  if (ray >= N_RAYS_C) return;
  const float dx = viewdirs[ray * 3], dy = viewdirs[ray * 3 + 1], dz = viewdirs[ray * 3 + 2];
  const float dn = sqrtf(dx * dx + dy * dy + dz * dz);
  float trans = 1.f, wacc = 0.f, c0 = 0.f, c1 = 0.f, c2 = 0.f;
  float tcur = tAll[ray * N_ALL_C];
  for (int i = 0; i < N_ALL_C; ++i) {
    const float tnext = (i < N_ALL_C - 1) ? tAll[ray * N_ALL_C + i + 1] : 0.f;
    const float dist  = ((i < N_ALL_C - 1) ? (tnext - tcur) : 1e10f) * dn;
    const int row = ray * N_ALL_C + i;
    const float a = 1.f - expf(-sigma[row] * dist);
    const float wgt = a * trans;
    trans *= (1.f - a + 1e-10f);
    c0 += wgt * rgb[row * 3 + 0];
    c1 += wgt * rgb[row * 3 + 1];
    c2 += wgt * rgb[row * 3 + 2];
    wacc += wgt;
    tcur = tnext;
  }
  const float bg = 1.f - wacc;   // white background
  out[ray * 3 + 0] = c0 + bg;
  out[ray * 3 + 1] = c1 + bg;
  out[ray * 3 + 2] = c2 + bg;
}

// ---------------- host driver ----------------------------------------------
extern "C" void kernel_launch(void* const* d_in, const int* in_sizes, int n_in,
                              void* d_out, int out_size, void* d_ws, size_t ws_size,
                              hipStream_t stream) {
  (void)in_sizes; (void)n_in; (void)out_size; (void)ws_size;
  const float* origins  = (const float*)d_in[0];
  const float* dirs     = (const float*)d_in[1];
  const float* viewdirs = (const float*)d_in[2];

  char* ws = (char*)d_ws;
  size_t off = 0;
  auto carve = [&](size_t bytes) -> char* {
    char* p = ws + off;
    off += (bytes + 255) & ~(size_t)255;
    return p;
  };

  // 12 matrices per MLP: 8 trunk, sigma, bottleneck, view, rgb
  static const int Kp[12]      = {64,256,256,256,256,320,256,256, 256,256,288,128};
  static const int din[12]     = {63,256,256,256,256,319,256,256, 256,256,283,128};
  static const int dout[12]    = {256,256,256,256,256,256,256,256, 1,256,128,3};
  static const int doutPad[12] = {256,256,256,256,256,256,256,256, 16,256,128,16};
  static const int wslot[12]   = {0,2,4,6,8,10,12,14, 16,18,20,22};

  _Float16* wtPtr[2][12];
  const float* bPtr[2][12];
  for (int m = 0; m < 2; ++m) {
    const int base = 3 + m * 24;     // params_coarse then params_fine, dict order
    for (int j = 0; j < 12; ++j) {
      const size_t halves = (size_t)Kp[j] * doutPad[j];
      wtPtr[m][j] = (_Float16*)carve(halves * 2);
      bPtr[m][j]  = (const float*)d_in[base + wslot[j] + 1];
      const float* W = (const float*)d_in[base + wslot[j]];
      const int total = Kp[j] * doutPad[j];
      convert_weight<<<(total + 255) / 256, 256, 0, stream>>>(
          W, wtPtr[m][j], din[j], dout[j], Kp[j], doutPad[j]);
    }
  }
  _Float16* vdenc = (_Float16*)carve((size_t)N_RAYS_C * 32 * 2);
  float* sigmaBuf = (float*)carve((size_t)N_RAYS_C * N_ALL_C * 4);
  float* rgbBuf   = (float*)carve((size_t)N_RAYS_C * N_ALL_C * 3 * 4);
  float* wBuf     = (float*)carve((size_t)N_RAYS_C * 64 * 4);
  float* tAll     = (float*)carve((size_t)N_RAYS_C * N_ALL_C * 4);

  vdenc_kernel<<<4, 256, 0, stream>>>(viewdirs, vdenc);

  MlpParams P[2];
  for (int m = 0; m < 2; ++m) {
    for (int i = 0; i < 8; ++i) { P[m].wt[i] = wtPtr[m][i]; P[m].bt[i] = bPtr[m][i]; }
    P[m].wsig  = wtPtr[m][8];  P[m].bsig  = bPtr[m][8];
    P[m].wbot  = wtPtr[m][9];  P[m].bbot  = bPtr[m][9];
    P[m].wview = wtPtr[m][10]; P[m].bview = bPtr[m][10];
    P[m].wrgb  = wtPtr[m][11]; P[m].brgb  = bPtr[m][11];
  }

  // coarse pass: 65536 samples / 64 rows per block = 1024 blocks (sigma only)
  fused_nerf_mlp<<<1024, 256, 0, stream>>>(P[0], origins, dirs, vdenc,
                                           nullptr, N_COARSE_C, sigmaBuf, rgbBuf, 0);
  render_coarse_weights<<<4, 256, 0, stream>>>(sigmaBuf, viewdirs, wBuf);
  pdf_merge_kernel<<<4, 256, 0, stream>>>(wBuf, tAll);

  // fine pass: 196608 samples / 64 = 3072 blocks (sigma + rgb)
  fused_nerf_mlp<<<3072, 256, 0, stream>>>(P[1], origins, dirs, vdenc,
                                           tAll, N_ALL_C, sigmaBuf, rgbBuf, 1);
  render_fine_kernel<<<4, 256, 0, stream>>>(sigmaBuf, rgbBuf, tAll, viewdirs,
                                            (float*)d_out);
}
